// CMIFE_1314259993166
// MI455X (gfx1250) — matrix operand order repaired and checked by reading, not compiled, not run on hardware
//
#include <hip/hip_runtime.h>
#include <cstdint>

#define BB   8
#define CC   256
#define HH   160
#define WW   160
#define HW   25600        // 160*160
#define MIDC 16
#define EPS  1e-5f

typedef __attribute__((ext_vector_type(8)))  float          v8f;
typedef __attribute__((ext_vector_type(16))) __bf16         v16bf;
typedef __attribute__((ext_vector_type(8)))  unsigned short v8u;
typedef __attribute__((ext_vector_type(16))) unsigned short v16u;

__device__ __forceinline__ unsigned short bf16_rne(float f) {
    unsigned u = __float_as_uint(f);
    u += 0x7FFFu + ((u >> 16) & 1u);
    return (unsigned short)(u >> 16);
}
__device__ __forceinline__ float sigmoidf_(float x) { return 1.f / (1.f + expf(-x)); }

// ---------------------------------------------------------------------------
// K1: align 1x1 conv as WMMA GEMM (bf16 hi/lo split, f32 accumulate) + BN + SiLU
// Y[b,co,p] = silu(bn(sum_ci W[co,ci]*X[b,ci,p]))
// Block: 256 thr = 8 waves. Tile 64(M=co) x 64(N=p). Wave: wm=wave&3 (16 rows),
// wn=wave>>2 (32 cols = two 16x16 subtiles). K chunked by 32.
// Software-pipelined: next chunk's global loads issue into registers while the
// current chunk's WMMAs run from LDS. Two accumulators per subtile (hi*hi and
// the two cross terms) break the WMMA RAW chain.
// ---------------------------------------------------------------------------
__global__ void __launch_bounds__(256)
k_align_gemm(const float* __restrict__ X, const float* __restrict__ Wt,
             const float* __restrict__ bg, const float* __restrict__ bb,
             const float* __restrict__ bm, const float* __restrict__ bv,
             float* __restrict__ Y)
{
    // stride 48 halves (96B): keeps fragment loads 16B/32B aligned, spreads banks
    __shared__ unsigned short sAhi[64 * 48], sAlo[64 * 48];
    __shared__ unsigned short sBhi[64 * 48], sBlo[64 * 48];
    __shared__ float sScale[64], sShift[64];

    const int tid  = threadIdx.x;
    const int lane = tid & 31, wave = tid >> 5;
    const int wm = wave & 3, wn = wave >> 2;
    const int g  = lane >> 4, ln = lane & 15;
    const int b   = blockIdx.z;
    const int co0 = blockIdx.y * 64;
    const int p0  = blockIdx.x * 64;

    if (tid < 64) {
        int co = co0 + tid;
        float sc = bg[co] * rsqrtf(bv[co] + EPS);
        sScale[tid] = sc;
        sShift[tid] = bb[co] - bm[co] * sc;
    }

    v8f aHH[2] = {};   // Ahi*Bhi
    v8f aLO[2] = {};   // Alo*Bhi + Ahi*Blo

    const int ciL = tid & 31, mrow = tid >> 5;   // W-chunk loader mapping
    const int nnL = tid & 63, krow = tid >> 6;   // X-chunk loader mapping
    const size_t xbase = (size_t)b * CC * HW + p0;

    // prologue: chunk 0 into registers
    float wv[8], xv[8];
#pragma unroll
    for (int i = 0; i < 8; ++i)
        wv[i] = Wt[(size_t)(co0 + mrow + 8 * i) * CC + ciL];
#pragma unroll
    for (int i = 0; i < 8; ++i)
        xv[i] = __builtin_nontemporal_load(&X[xbase + (size_t)(krow + 4 * i) * HW + nnL]);

    for (int k0 = 0; k0 < CC; k0 += 32) {
        __syncthreads();   // previous chunk's WMMA reads done before LDS overwrite
        // stage registers -> LDS as bf16 hi/lo (exact hi via truncation)
#pragma unroll
        for (int i = 0; i < 8; ++i) {
            int mm = mrow + 8 * i;
            float v = wv[i];
            unsigned u = __float_as_uint(v);
            float hi = __uint_as_float(u & 0xFFFF0000u);
            sAhi[mm * 48 + ciL] = (unsigned short)(u >> 16);
            sAlo[mm * 48 + ciL] = bf16_rne(v - hi);       // v-hi exact in f32
        }
#pragma unroll
        for (int i = 0; i < 8; ++i) {
            int kk = krow + 4 * i;
            float v = xv[i];
            unsigned u = __float_as_uint(v);
            float hi = __uint_as_float(u & 0xFFFF0000u);
            sBhi[nnL * 48 + kk] = (unsigned short)(u >> 16);
            sBlo[nnL * 48 + kk] = bf16_rne(v - hi);
        }
        __syncthreads();

        // issue next chunk's global loads; latency hidden under WMMA below
        if (k0 + 32 < CC) {
            const int kn = k0 + 32;
#pragma unroll
            for (int i = 0; i < 8; ++i)
                wv[i] = Wt[(size_t)(co0 + mrow + 8 * i) * CC + kn + ciL];
#pragma unroll
            for (int i = 0; i < 8; ++i)
                xv[i] = __builtin_nontemporal_load(
                    &X[xbase + (size_t)(kn + krow + 4 * i) * HW + nnL]);
        }

        // A fragment (16x32 bf16): lane holds row m=ln; element e -> K=16g+e
        // (contiguous, matching B)
        const int mr = 16 * wm + ln;
        v8u a0h = *(const v8u*)&sAhi[mr * 48 + 16 * g];
        v8u a1h = *(const v8u*)&sAhi[mr * 48 + 8 + 16 * g];
        v8u a0l = *(const v8u*)&sAlo[mr * 48 + 16 * g];
        v8u a1l = *(const v8u*)&sAlo[mr * 48 + 8 + 16 * g];
        v16u th, tl;
#pragma unroll
        for (int i = 0; i < 8; ++i) {
            th[i] = a0h[i]; th[8 + i] = a1h[i];
            tl[i] = a0l[i]; tl[8 + i] = a1l[i];
        }
        v16bf Ahi = __builtin_bit_cast(v16bf, th);
        v16bf Alo = __builtin_bit_cast(v16bf, tl);

#pragma unroll
        for (int sub = 0; sub < 2; ++sub) {
            // B fragment (32x16 bf16): lane holds col n=ln, K = 16g + e (contiguous)
            int nc = 32 * wn + 16 * sub + ln;
            v16bf Bhi = __builtin_bit_cast(v16bf, *(const v16u*)&sBhi[nc * 48 + 16 * g]);
            v16bf Blo = __builtin_bit_cast(v16bf, *(const v16u*)&sBlo[nc * 48 + 16 * g]);
            // fp32-ish product: (hi*hi) and (lo*hi + hi*lo) in separate chains
            aHH[sub] = __builtin_amdgcn_wmma_f32_16x16x32_bf16(false, Ahi, false, Bhi,
                                                               (short)0, aHH[sub], false, false);
            aLO[sub] = __builtin_amdgcn_wmma_f32_16x16x32_bf16(false, Alo, false, Bhi,
                                                               (short)0, aLO[sub], false, false);
            aLO[sub] = __builtin_amdgcn_wmma_f32_16x16x32_bf16(false, Ahi, false, Blo,
                                                               (short)0, aLO[sub], false, false);
        }
    }

    // D layout: VGPR r -> M = 8*g + r, N = lane%16. Combine chains, BN + SiLU, store.
#pragma unroll
    for (int sub = 0; sub < 2; ++sub) {
#pragma unroll
        for (int r = 0; r < 8; ++r) {
            int coL = 16 * wm + 8 * g + r;
            int p = p0 + 32 * wn + 16 * sub + ln;
            float v = (aHH[sub][r] + aLO[sub][r]) * sScale[coL] + sShift[coL];
            v = v * sigmoidf_(v);
            Y[((size_t)(b * CC + co0 + coL)) * HW + p] = v;
        }
    }
}

// ---------------------------------------------------------------------------
// K2: per-(b,c) global mean, global max, and 4x4 local-pool means (40x40 cells)
// Block per (b,c); 16 threads per cell (static cell assignment).
// ---------------------------------------------------------------------------
__global__ void __launch_bounds__(256)
k_pool(const float* __restrict__ Y, float* __restrict__ gmean,
       float* __restrict__ gmax, float* __restrict__ lmean)
{
    __shared__ float red[256];
    const int tid = threadIdx.x;
    const int bc = blockIdx.x;
    const float* yb = Y + (size_t)bc * HW;
    const int cell = tid >> 4, lane = tid & 15;
    const int cy = cell >> 2, cx = cell & 3;

    float s = 0.f, mx = -3.4e38f;
    for (int j = lane; j < 1600; j += 16) {
        int rr = cy * 40 + j / 40;
        int cc = cx * 40 + j % 40;
        float v = yb[rr * WW + cc];
        s += v; mx = fmaxf(mx, v);
    }
    red[tid] = s; __syncthreads();
    for (int st = 8; st > 0; st >>= 1) {
        if (lane < st) red[tid] += red[tid + st];
        __syncthreads();
    }
    if (tid < 16) lmean[(size_t)bc * 16 + tid] = red[tid * 16] * (1.f / 1600.f);
    if (tid == 0) {
        float t = 0.f;
        for (int c = 0; c < 16; ++c) t += red[c * 16];
        gmean[bc] = t * (1.f / (float)HW);
    }
    __syncthreads();
    red[tid] = mx; __syncthreads();
    for (int st = 128; st > 0; st >>= 1) {
        if (tid < st) red[tid] = fmaxf(red[tid], red[tid + st]);
        __syncthreads();
    }
    if (tid == 0) gmax[bc] = red[0];
}

// ---------------------------------------------------------------------------
// K3: channel attention. One block per batch (320 threads).
// local positional mean commutes with the 2nd 1x1 conv -> average hidden first.
// ---------------------------------------------------------------------------
__global__ void __launch_bounds__(320)
k_chattn(const float* __restrict__ gmean, const float* __restrict__ gmax,
         const float* __restrict__ lmean, const float* __restrict__ w1,
         const float* __restrict__ w2, const float* __restrict__ lw1,
         const float* __restrict__ lw2, const float* __restrict__ fusion,
         float* __restrict__ ca)
{
    __shared__ float hidA[16], hidM[16], hidL[16][16], hbar[16];
    const int tid = threadIdx.x;
    const int b = blockIdx.x;

    if (tid < 16) {
        float a = 0.f;
        for (int c = 0; c < CC; ++c) a += w1[tid * CC + c] * gmean[b * CC + c];
        hidA[tid] = fmaxf(a, 0.f);
    } else if (tid < 32) {
        int h = tid - 16; float a = 0.f;
        for (int c = 0; c < CC; ++c) a += w1[h * CC + c] * gmax[b * CC + c];
        hidM[h] = fmaxf(a, 0.f);
    } else if (tid < 288) {
        int idx = tid - 32, q = idx >> 4, h = idx & 15;
        float a = 0.f;
        for (int c = 0; c < CC; ++c)
            a += lw1[h * CC + c] * lmean[((size_t)(b * CC + c)) * 16 + q];
        hidL[q][h] = fmaxf(a, 0.f);
    }
    __syncthreads();
    if (tid < 16) {
        float a = 0.f;
        for (int q = 0; q < 16; ++q) a += hidL[q][tid];
        hbar[tid] = a * (1.f / 16.f);
    }
    __syncthreads();
    if (tid < CC) {
        float alpha = sigmoidf_(fusion[0]);
        float ga = 0.f, gm = 0.f, la = 0.f;
        for (int h = 0; h < 16; ++h) {
            float w = w2[tid * 16 + h];
            ga += w * hidA[h];
            gm += w * hidM[h];
            la += lw2[tid * 16 + h] * hbar[h];
        }
        ca[b * CC + tid] = sigmoidf_(alpha * (ga + gm) + (1.f - alpha) * la);
    }
}

// ---------------------------------------------------------------------------
// K4: fold 98->2 channel mean into conv2 weights/bias (mean commutes with conv)
// ---------------------------------------------------------------------------
__global__ void k_fold(const float* __restrict__ w2, const float* __restrict__ b2,
                       float* __restrict__ effw, float* __restrict__ effb)
{
    int t = blockIdx.x * blockDim.x + threadIdx.x;
    if (t < 288) {
        int d = t / 144, r = t % 144;
        float a = 0.f;
        for (int j = 0; j < 49; ++j) a += w2[(size_t)(d * 49 + j) * 144 + r];
        effw[t] = a * (1.f / 49.f);
    } else if (t < 290) {
        int d = t - 288; float a = 0.f;
        for (int j = 0; j < 49; ++j) a += b2[d * 49 + j];
        effb[d] = a * (1.f / 49.f);
    }
}

// ---------------------------------------------------------------------------
// K5: sf = [mean_c(ca*y), max_c(ca*y)]; ca applied on the fly (x2 never stored)
// ---------------------------------------------------------------------------
__global__ void __launch_bounds__(256)
k_sf(const float* __restrict__ Y, const float* __restrict__ ca,
     float* __restrict__ sf)
{
    __shared__ float sca[CC];
    const int tid = threadIdx.x;
    const int b = blockIdx.x / 100;
    const int p = (blockIdx.x % 100) * 256 + tid;
    sca[tid] = ca[b * CC + tid];
    __syncthreads();
    const float* yb = Y + (size_t)b * CC * HW + p;
    float s = 0.f, mx = -3.4e38f;
    for (int c = 0; c < CC; ++c) {
        float v = sca[c] * yb[(size_t)c * HW];
        s += v; mx = fmaxf(mx, v);
    }
    sf[((size_t)b * 2 + 0) * HW + p] = s * (1.f / (float)CC);
    sf[((size_t)b * 2 + 1) * HW + p] = mx;
}

// ---------------------------------------------------------------------------
// K6: offset conv1 3x3 (2->16) + BN + ReLU
// ---------------------------------------------------------------------------
__global__ void __launch_bounds__(256)
k_off1(const float* __restrict__ sf, const float* __restrict__ w1,
       const float* __restrict__ g, const float* __restrict__ bt,
       const float* __restrict__ m, const float* __restrict__ vv,
       float* __restrict__ T)
{
    __shared__ float sw[288], ssc[16], ssh[16];
    const int tid = threadIdx.x;
    if (tid < 288) sw[tid] = w1[tid];
    if (tid < 16) {
        float sc = g[tid] * rsqrtf(vv[tid] + EPS);
        ssc[tid] = sc; ssh[tid] = bt[tid] - m[tid] * sc;
    }
    __syncthreads();
    const int b = blockIdx.x / 100;
    const int p = (blockIdx.x % 100) * 256 + tid;
    const int r = p / WW, cl = p % WW;

    float nb[2][9];
    for (int ic = 0; ic < 2; ++ic)
        for (int kh = 0; kh < 3; ++kh)
            for (int kw = 0; kw < 3; ++kw) {
                int rr = r + kh - 1, cc = cl + kw - 1;
                nb[ic][kh * 3 + kw] = (rr >= 0 && rr < HH && cc >= 0 && cc < WW)
                    ? sf[((size_t)b * 2 + ic) * HW + rr * WW + cc] : 0.f;
            }
    for (int oc = 0; oc < 16; ++oc) {
        float a = 0.f;
        for (int ic = 0; ic < 2; ++ic)
            for (int k = 0; k < 9; ++k)
                a += sw[oc * 18 + ic * 9 + k] * nb[ic][k];
        a = a * ssc[oc] + ssh[oc];
        T[((size_t)(b * 16 + oc)) * HW + p] = fmaxf(a, 0.f);
    }
}

// ---------------------------------------------------------------------------
// K7: offset conv2 3x3 (16->2, folded) + bias, then tanh*0.5
// ---------------------------------------------------------------------------
__global__ void __launch_bounds__(256)
k_off2(const float* __restrict__ T, const float* __restrict__ effw,
       const float* __restrict__ effb, float* __restrict__ off)
{
    __shared__ float sw[288], sb[2];
    const int tid = threadIdx.x;
    if (tid < 288) sw[tid] = effw[tid];
    if (tid < 2) sb[tid] = effb[tid];
    __syncthreads();
    const int b = blockIdx.x / 100;
    const int p = (blockIdx.x % 100) * 256 + tid;
    const int r = p / WW, cl = p % WW;

    float a0 = sb[0], a1 = sb[1];
    for (int ic = 0; ic < 16; ++ic) {
        const float* tb = T + ((size_t)(b * 16 + ic)) * HW;
        for (int kh = 0; kh < 3; ++kh) {
            int rr = r + kh - 1;
            if (rr < 0 || rr >= HH) continue;
            for (int kw = 0; kw < 3; ++kw) {
                int cc = cl + kw - 1;
                if (cc < 0 || cc >= WW) continue;
                float v = tb[rr * WW + cc];
                a0 += sw[ic * 9 + kh * 3 + kw] * v;
                a1 += sw[144 + ic * 9 + kh * 3 + kw] * v;
            }
        }
    }
    off[((size_t)b * 2 + 0) * HW + p] = tanhf(a0) * 0.5f;
    off[((size_t)b * 2 + 1) * HW + p] = tanhf(a1) * 0.5f;
}

// ---------------------------------------------------------------------------
// K8: bilinear grid-sample of sf at base+offset (zeros padding, align_corners=F)
// ---------------------------------------------------------------------------
__global__ void __launch_bounds__(256)
k_sample(const float* __restrict__ sf, const float* __restrict__ off,
         float* __restrict__ smp)
{
    const int tid = threadIdx.x;
    const int b = blockIdx.x / 100;
    const int p = (blockIdx.x % 100) * 256 + tid;
    const int r = p / WW, cl = p % WW;

    float gx = -1.f + (2.f / 159.f) * (float)cl + off[((size_t)b * 2 + 0) * HW + p];
    float gy = -1.f + (2.f / 159.f) * (float)r  + off[((size_t)b * 2 + 1) * HW + p];
    gx = fminf(fmaxf(gx, -1.f), 1.f);
    gy = fminf(fmaxf(gy, -1.f), 1.f);
    float fx = (gx + 1.f) * 80.f - 0.5f;
    float fy = (gy + 1.f) * 80.f - 0.5f;
    float x0f = floorf(fx), y0f = floorf(fy);
    float wx = fx - x0f, wy = fy - y0f;
    int x0 = (int)x0f, y0 = (int)y0f;
    int x1 = x0 + 1, y1 = y0 + 1;

    for (int ch = 0; ch < 2; ++ch) {
        const float* fb = sf + ((size_t)b * 2 + ch) * HW;
        float s00 = (x0 >= 0 && x0 < WW && y0 >= 0 && y0 < HH) ? fb[y0 * WW + x0] : 0.f;
        float s01 = (x1 >= 0 && x1 < WW && y0 >= 0 && y0 < HH) ? fb[y0 * WW + x1] : 0.f;
        float s10 = (x0 >= 0 && x0 < WW && y1 >= 0 && y1 < HH) ? fb[y1 * WW + x0] : 0.f;
        float s11 = (x1 >= 0 && x1 < WW && y1 >= 0 && y1 < HH) ? fb[y1 * WW + x1] : 0.f;
        smp[((size_t)b * 2 + ch) * HW + p] =
            s00 * (1.f - wx) * (1.f - wy) + s01 * wx * (1.f - wy)
          + s10 * (1.f - wx) * wy        + s11 * wx * wy;
    }
}

// ---------------------------------------------------------------------------
// K9: 7x7 conv (2->1) pad 3 + sigmoid -> sa
// ---------------------------------------------------------------------------
__global__ void __launch_bounds__(256)
k_attn(const float* __restrict__ smp, const float* __restrict__ aw,
       float* __restrict__ sa)
{
    __shared__ float sw[98];
    const int tid = threadIdx.x;
    if (tid < 98) sw[tid] = aw[tid];
    __syncthreads();
    const int b = blockIdx.x / 100;
    const int p = (blockIdx.x % 100) * 256 + tid;
    const int r = p / WW, cl = p % WW;

    float a = 0.f;
    for (int ic = 0; ic < 2; ++ic) {
        const float* fb = smp + ((size_t)b * 2 + ic) * HW;
        for (int kh = 0; kh < 7; ++kh) {
            int rr = r + kh - 3;
            if (rr < 0 || rr >= HH) continue;
            for (int kw = 0; kw < 7; ++kw) {
                int cc = cl + kw - 3;
                if (cc < 0 || cc >= WW) continue;
                a += sw[ic * 49 + kh * 7 + kw] * fb[rr * WW + cc];
            }
        }
    }
    sa[(size_t)b * HW + p] = sigmoidf_(a);
}

// ---------------------------------------------------------------------------
// K10: out = sa * ca * y  (out is write-only -> non-temporal store)
// ---------------------------------------------------------------------------
__global__ void __launch_bounds__(256)
k_final(const float* __restrict__ Y, const float* __restrict__ ca,
        const float* __restrict__ sa, float* __restrict__ out)
{
    size_t i = (size_t)blockIdx.x * 256 + threadIdx.x;
    int p = (int)(i % HW);
    int c = (int)((i / HW) % CC);
    int b = (int)(i / ((size_t)HW * CC));
    float v = sa[(size_t)b * HW + p] * ca[b * CC + c] * Y[i];
    __builtin_nontemporal_store(v, &out[i]);
}

// ---------------------------------------------------------------------------
extern "C" void kernel_launch(void* const* d_in, const int* in_sizes, int n_in,
                              void* d_out, int out_size, void* d_ws, size_t ws_size,
                              hipStream_t stream)
{
    const float* x       = (const float*)d_in[0];
    const float* align_w = (const float*)d_in[1];
    const float* align_g = (const float*)d_in[2];
    const float* align_b = (const float*)d_in[3];
    const float* align_m = (const float*)d_in[4];
    const float* align_v = (const float*)d_in[5];
    const float* mlp_w1  = (const float*)d_in[6];
    const float* mlp_w2  = (const float*)d_in[7];
    const float* loc_w1  = (const float*)d_in[8];
    const float* loc_w2  = (const float*)d_in[9];
    const float* fusion  = (const float*)d_in[10];
    const float* off_w1  = (const float*)d_in[11];
    const float* off_g   = (const float*)d_in[12];
    const float* off_bt  = (const float*)d_in[13];
    const float* off_m   = (const float*)d_in[14];
    const float* off_v   = (const float*)d_in[15];
    const float* off_w2  = (const float*)d_in[16];
    const float* off_b2  = (const float*)d_in[17];
    const float* attn_w  = (const float*)d_in[18];
    float* out = (float*)d_out;

    // workspace layout (floats)
    float* ws = (float*)d_ws;
    size_t o = 0;
    float* Y     = ws + o; o += (size_t)BB * CC * HW;   // 52,428,800
    float* sf    = ws + o; o += (size_t)BB * 2 * HW;    // 409,600
    float* T     = ws + o; o += (size_t)BB * 16 * HW;   // 3,276,800
    float* offb  = ws + o; o += (size_t)BB * 2 * HW;    // 409,600
    float* smp   = ws + o; o += (size_t)BB * 2 * HW;    // 409,600
    float* sa    = ws + o; o += (size_t)BB * HW;        // 204,800
    float* gmean = ws + o; o += (size_t)BB * CC;
    float* gmax  = ws + o; o += (size_t)BB * CC;
    float* lmean = ws + o; o += (size_t)BB * CC * 16;
    float* ca    = ws + o; o += (size_t)BB * CC;
    float* effw  = ws + o; o += 288;
    float* effb  = ws + o; o += 2;

    // 1) align 1x1 conv (WMMA GEMM) + BN + SiLU
    k_align_gemm<<<dim3(HW / 64, CC / 64, BB), 256, 0, stream>>>(
        x, align_w, align_g, align_b, align_m, align_v, Y);

    // 2) pools
    k_pool<<<BB * CC, 256, 0, stream>>>(Y, gmean, gmax, lmean);

    // 3) channel attention
    k_chattn<<<BB, 320, 0, stream>>>(gmean, gmax, lmean, mlp_w1, mlp_w2,
                                     loc_w1, loc_w2, fusion, ca);

    // 4) fold 98->2 offset-channel mean into conv2 weights (independent)
    k_fold<<<1, 320, 0, stream>>>(off_w2, off_b2, effw, effb);

    // 5) sf = channel mean/max of ca*y
    k_sf<<<BB * 100, 256, 0, stream>>>(Y, ca, sf);

    // 6) offset conv1 + BN + ReLU
    k_off1<<<BB * 100, 256, 0, stream>>>(sf, off_w1, off_g, off_bt, off_m, off_v, T);

    // 7) offset conv2 (folded) + tanh*0.5
    k_off2<<<BB * 100, 256, 0, stream>>>(T, effw, effb, offb);

    // 8) grid sample
    k_sample<<<BB * 100, 256, 0, stream>>>(sf, offb, smp);

    // 9) 7x7 conv + sigmoid
    k_attn<<<BB * 100, 256, 0, stream>>>(smp, attn_w, sa);

    // 10) out = sa * ca * y
    k_final<<<(unsigned)(((size_t)BB * CC * HW) / 256), 256, 0, stream>>>(Y, ca, sa, out);
}
